// DeepRouterMoeForward_94489280668
// MI455X (gfx1250) — compile-verified, matrix operand order checked by reading, not run on hardware
//
#include <hip/hip_runtime.h>

// ---------------------------------------------------------------------------
// DeepRouterMoeForward for MI455X (gfx1250): LDS-tiled bf16 WMMA formulation.
//   T=2048 tokens, H=2048, O=2048, E=8.
//   Pass 1: convert h / router_w / expert_w to bf16 in workspace.
//   Pass 2: router GEMM [T x (O*E)] with fused softmax/top2 -> c[T,8] (atomics)
//   Pass 3: A'[t, e*H+h] = c[t,e]*h[t,h]  (bf16)
//   Pass 4: final = A' @ B'   with B'[(e,h),o] = expert_w[e,o,h], + bias term
// GEMM: block tile 128x128, K-chunk 32, TRIPLE-buffered LDS staged with
// gfx1250 async global->LDS copies; two chunks kept in flight (partial
// s_wait_asynccnt <= 4) so load latency spans two compute phases.
// ---------------------------------------------------------------------------

typedef __attribute__((ext_vector_type(16))) __bf16 v16bf;
typedef __attribute__((ext_vector_type(8)))  __bf16 v8bf;
typedef __attribute__((ext_vector_type(8)))  float  v8f;
typedef __attribute__((ext_vector_type(4)))  int    v4i;
typedef __attribute__((address_space(1))) v4i v4i_g;   // global int4
typedef __attribute__((address_space(3))) v4i v4i_l;   // LDS int4

#define T_TOK 2048
#define HDIM  2048
#define ODIM  2048
#define NEXP  8
#define OE    (ODIM * NEXP)   // 16384 router output columns
#define KE    (NEXP * HDIM)   // 16384 fused expert K dimension
#define KC    32              // K-chunk per LDS stage
#define LDST  40              // LDS row stride in bf16 (80B: 16B aligned, bank-skewed)

#if defined(__has_builtin)
#if __has_builtin(__builtin_amdgcn_global_load_async_to_lds_b128)
#define HAVE_ASYNC_COPY 1
#endif
#endif

#ifdef HAVE_ASYNC_COPY
#if __has_builtin(__builtin_amdgcn_s_wait_asynccnt)
#define WAIT_ASYNC(n) __builtin_amdgcn_s_wait_asynccnt(n)
#else
#define WAIT_ASYNC(n) asm volatile("s_wait_asynccnt %0" ::"i"(n) : "memory")
#endif
#endif

union FragBF {
  v16bf v;
  v8bf  h[2];
};

__device__ inline void argmax8(float& v, int& i) {
#pragma unroll
  for (int m = 1; m < 8; m <<= 1) {
    float ov = __shfl_xor(v, m, 32);
    int   oi = __shfl_xor(i, m, 32);
    if (ov > v || (ov == v && oi < i)) { v = ov; i = oi; }
  }
}

// ---------------------------------------------------------------------------
__global__ __launch_bounds__(256) void cvt_f32_bf16(const float* __restrict__ in,
                                                    __bf16* __restrict__ out,
                                                    long n) {
  long i = ((long)blockIdx.x * blockDim.x + threadIdx.x) * 4;
  if (i + 3 < n) {
    const float4 v = *(const float4*)(in + i);
    out[i + 0] = (__bf16)v.x;
    out[i + 1] = (__bf16)v.y;
    out[i + 2] = (__bf16)v.z;
    out[i + 3] = (__bf16)v.w;
  }
}

// ---------------------------------------------------------------------------
// A'[t][e*H + h] = bf16(c[t,e] * h[t,h])
__global__ __launch_bounds__(256) void build_scaled_A(const float* __restrict__ hf,
                                                      const float* __restrict__ cbuf,
                                                      __bf16* __restrict__ out) {
  long i   = ((long)blockIdx.x * blockDim.x + threadIdx.x) * 4;  // over T*KE
  long t   = i / KE;
  long rem = i - t * KE;
  int  e   = (int)(rem >> 11);        // /HDIM
  long h   = rem & (HDIM - 1);
  float s  = cbuf[t * NEXP + e];
  const float4 v = *(const float4*)(hf + t * HDIM + h);
  out[i + 0] = (__bf16)(s * v.x);
  out[i + 1] = (__bf16)(s * v.y);
  out[i + 2] = (__bf16)(s * v.z);
  out[i + 3] = (__bf16)(s * v.w);
}

// ---------------------------------------------------------------------------
// Shared LDS-tiled GEMM for both passes.
//   ROUTER=true : C = A[T,H] x Bw[OE,H]^T, fused softmax/top2 -> cbuf (atomic)
//   ROUTER=false: C = A[T,KE] x B'(expert_w), + bias -> out
template <int KTOT, bool ROUTER>
__global__ __launch_bounds__(256) void moe_gemm(
    const __bf16* __restrict__ A,
    const __bf16* __restrict__ Bw,
    const float*  __restrict__ rb,    // router bias [OE]        (ROUTER)
    const float*  __restrict__ eb,    // expert bias [E,O]       (!ROUTER)
    float* __restrict__ cbuf,         // [T,NEXP] out(ROUTER) / in(!ROUTER)
    float* __restrict__ out) {        // [T,O]                   (!ROUTER)
  __shared__ __bf16 As[3][128 * LDST];   // 3 x 10KB
  __shared__ __bf16 Bs[3][128 * LDST];   // 3 x 10KB  (60KB total)

  const int tid  = threadIdx.x;
  const int wave = tid >> 5;
  const int lane = tid & 31;
  const int ln   = lane & 15;
  const int hi2  = lane >> 4;
  const int wm   = wave >> 2;   // 0..1 : which 64-row half of the block tile
  const int wn   = wave & 3;    // 0..3 : which 32-col quarter
  const int bm0  = blockIdx.y * 128;
  const int bn0  = blockIdx.x * 128;

  // Staging: 128 rows x 32 k (64B) per matrix = 512 x 16B transfers; 2/thread.
  auto a_gptr = [&](int r, int k, int s) -> const __bf16* {
    return A + (size_t)(bm0 + r) * KTOT + k + s * 8;
  };
  auto b_gptr = [&](int r, int k, int s) -> const __bf16* {
    if (ROUTER) {
      return Bw + (size_t)(bn0 + r) * HDIM + k + s * 8;
    } else {
      const int e = k >> 11;                        // k / HDIM
      const int h = (k & (HDIM - 1)) + s * 8;
      return Bw + (((size_t)e * ODIM + (bn0 + r)) << 11) + h;
    }
  };

  v8f acc[4][2] = {};

  auto compute = [&](int cur) {
    v16bf af[4], bfr[2];
#pragma unroll
    for (int mf = 0; mf < 4; ++mf) {
      const __bf16* p = &As[cur][(wm * 64 + mf * 16 + ln) * LDST + 8 * hi2];
      FragBF f; f.h[0] = *(const v8bf*)p; f.h[1] = *(const v8bf*)(p + 16);
      af[mf] = f.v;
    }
#pragma unroll
    for (int nf = 0; nf < 2; ++nf) {
      const __bf16* p = &Bs[cur][(wn * 32 + nf * 16 + ln) * LDST + 8 * hi2];
      FragBF f; f.h[0] = *(const v8bf*)p; f.h[1] = *(const v8bf*)(p + 16);
      bfr[nf] = f.v;
    }
#pragma unroll
    for (int mf = 0; mf < 4; ++mf)
#pragma unroll
      for (int nf = 0; nf < 2; ++nf)
        acc[mf][nf] = __builtin_amdgcn_wmma_f32_16x16x32_bf16(
            false, af[mf], false, bfr[nf], (short)0, acc[mf][nf], false, false);
  };

  constexpr int NK = KTOT / KC;

#ifdef HAVE_ASYNC_COPY
  // -------- gfx1250 async global->LDS path, 2 chunks in flight --------
  // Each chunk = 4 async-load instructions per wave; async loads complete
  // in order, so s_wait_asynccnt<=4 guarantees the older chunk has landed.
  auto stage_async = [&](int buf, int k0) {
#pragma unroll
    for (int it = 0; it < 2; ++it) {
      const int t = tid + it * 256, r = t >> 2, s = t & 3;
      __builtin_amdgcn_global_load_async_to_lds_b128(
          (v4i_g*)a_gptr(r, k0, s),
          (v4i_l*)&As[buf][r * LDST + s * 8], 0, 0);
      __builtin_amdgcn_global_load_async_to_lds_b128(
          (v4i_g*)b_gptr(r, k0, s),
          (v4i_l*)&Bs[buf][r * LDST + s * 8], 0, 0);
    }
  };

  stage_async(0, 0);
  if (NK > 1) stage_async(1, KC);
  WAIT_ASYNC(4);          // chunk 0 landed; chunk 1 still in flight
  __syncthreads();
  for (int kc = 0; kc < NK; ++kc) {
    const int cur = kc % 3;
    if (kc + 2 < NK) stage_async((kc + 2) % 3, (kc + 2) * KC);
    if (kc + 3 < NK) __builtin_prefetch(a_gptr(tid >> 2, (kc + 3) * KC, tid & 3), 0, 0);
    compute(cur);
    if (kc + 1 < NK) {
      if (kc + 2 < NK) { WAIT_ASYNC(4); }   // next chunk landed, newest flying
      else             { WAIT_ASYNC(0); }   // drain tail
    }
    __syncthreads();
  }
#else
  // -------- fallback: register-staged, software-pipelined, 2 buffers ------
  {
#pragma unroll
    for (int it = 0; it < 2; ++it) {
      const int t = tid + it * 256, r = t >> 2, s = t & 3;
      *(uint4*)&As[0][r * LDST + s * 8] = *(const uint4*)a_gptr(r, 0, s);
      *(uint4*)&Bs[0][r * LDST + s * 8] = *(const uint4*)b_gptr(r, 0, s);
    }
  }
  __syncthreads();
  for (int kc = 0; kc < NK; ++kc) {
    const int cur = kc & 1, nxt = cur ^ 1;
    uint4 rga[2], rgb[2];
    if (kc + 1 < NK) {
      const int k0n = (kc + 1) * KC;
#pragma unroll
      for (int it = 0; it < 2; ++it) {
        const int t = tid + it * 256, r = t >> 2, s = t & 3;
        rga[it] = *(const uint4*)a_gptr(r, k0n, s);
        rgb[it] = *(const uint4*)b_gptr(r, k0n, s);
      }
    }
    if (kc + 2 < NK) __builtin_prefetch(a_gptr(tid >> 2, (kc + 2) * KC, tid & 3), 0, 0);
    compute(cur);
    if (kc + 1 < NK) {
#pragma unroll
      for (int it = 0; it < 2; ++it) {
        const int t = tid + it * 256, r = t >> 2, s = t & 3;
        *(uint4*)&As[nxt][r * LDST + s * 8] = rga[it];
        *(uint4*)&Bs[nxt][r * LDST + s * 8] = rgb[it];
      }
    }
    __syncthreads();
  }
#endif

  // ------------------------- epilogues -------------------------
  if (ROUTER) {
    // Fused per-(token,feature) softmax over E=8 + top-2 renorm.
    // Column group of 8 == 8-lane shuffle subgroup (masks 1/2/4 stay inside).
    float cacc[4][8] = {};
    const int e_lane = lane & 7;
#pragma unroll
    for (int mf = 0; mf < 4; ++mf) {
#pragma unroll
      for (int nf = 0; nf < 2; ++nf) {
        const int   ncol = bn0 + wn * 32 + nf * 16 + ln;
        const float rbv  = rb[ncol];
#pragma unroll
        for (int j = 0; j < 8; ++j) {
          float v = acc[mf][nf][j] + rbv;
          float gmax = v;
          gmax = fmaxf(gmax, __shfl_xor(gmax, 1, 32));
          gmax = fmaxf(gmax, __shfl_xor(gmax, 2, 32));
          gmax = fmaxf(gmax, __shfl_xor(gmax, 4, 32));
          float p = __expf(v - gmax);   // softmax Z cancels in top-2 renorm

          float v1 = p;  int i1 = e_lane;
          argmax8(v1, i1);
          float v2 = (e_lane == i1) ? -1.0f : p;  // probs are > 0
          int   i2 = e_lane;
          argmax8(v2, i2);

          const float tot = v1 + v2;
          const float w   = (e_lane == i1) ? (v1 / tot)
                           : ((e_lane == i2) ? (v2 / tot) : 0.0f);
          cacc[mf][j] += w;
        }
      }
    }
#pragma unroll
    for (int mf = 0; mf < 4; ++mf)
#pragma unroll
      for (int j = 0; j < 8; ++j) {
        const int row = bm0 + wm * 64 + mf * 16 + j + 8 * hi2;
        atomicAdd(&cbuf[(size_t)row * NEXP + e_lane], cacc[mf][j]);
      }
  } else {
#pragma unroll
    for (int mf = 0; mf < 4; ++mf) {
#pragma unroll
      for (int nf = 0; nf < 2; ++nf) {
        const int ncol = bn0 + wn * 32 + nf * 16 + ln;
#pragma unroll
        for (int j = 0; j < 8; ++j) {
          const int row = bm0 + wm * 64 + mf * 16 + j + 8 * hi2;
          float bias = 0.0f;
#pragma unroll
          for (int e = 0; e < NEXP; ++e)
            bias += cbuf[(size_t)row * NEXP + e] * eb[(size_t)e * ODIM + ncol];
          out[(size_t)row * ODIM + ncol] = acc[mf][nf][j] + bias;
        }
      }
    }
  }
}

// ---------------------------------------------------------------------------
extern "C" void kernel_launch(void* const* d_in, const int* in_sizes, int n_in,
                              void* d_out, int out_size, void* d_ws, size_t ws_size,
                              hipStream_t stream) {
  (void)in_sizes; (void)n_in; (void)out_size; (void)ws_size;
  const float* hf  = (const float*)d_in[0];  // [T, H]
  const float* rw  = (const float*)d_in[1];  // [OE, H]
  const float* rbi = (const float*)d_in[2];  // [OE]
  const float* ew  = (const float*)d_in[3];  // [E, O, H]
  const float* eb  = (const float*)d_in[4];  // [E, O]
  float* out = (float*)d_out;                // [T, O]

  char* ws = (char*)d_ws;
  const size_t MB = 1024 * 1024;
  __bf16* hbf  = (__bf16*)(ws);                         //   8 MB [T,H]
  __bf16* rwbf = (__bf16*)(ws + 8 * MB);                //  64 MB [OE,H]
  __bf16* ewbf = (__bf16*)(ws + 72 * MB);               //  64 MB [E,O,H]
  __bf16* chbf = (__bf16*)(ws + 136 * MB);              //  64 MB [T,KE]
  float*  cbuf = (float*)(ws + 200 * MB);               //  64 KB [T,NEXP]

  const long nh   = (long)T_TOK * HDIM;         // 4 Mi
  const long nrw  = (long)OE * HDIM;            // 32 Mi
  const long n_ew = (long)NEXP * ODIM * HDIM;   // 32 Mi

  cvt_f32_bf16<<<(int)(nh / 1024), 256, 0, stream>>>(hf, hbf, nh);
  cvt_f32_bf16<<<(int)(nrw / 1024), 256, 0, stream>>>(rw, rwbf, nrw);
  cvt_f32_bf16<<<(int)(n_ew / 1024), 256, 0, stream>>>(ew, ewbf, n_ew);

  (void)hipMemsetAsync(cbuf, 0, (size_t)T_TOK * NEXP * sizeof(float), stream);

  moe_gemm<HDIM, true><<<dim3(OE / 128, T_TOK / 128), 256, 0, stream>>>(
      hbf, rwbf, rbi, nullptr, cbuf, nullptr);

  build_scaled_A<<<(int)(((long)T_TOK * KE) / 1024), 256, 0, stream>>>(hf, cbuf, chbf);

  moe_gemm<KE, false><<<dim3(ODIM / 128, T_TOK / 128), 256, 0, stream>>>(
      chbf, ewbf, nullptr, eb, cbuf, out);
}
